// EnhancedTransformerBlock_28269474743022
// MI455X (gfx1250) — matrix operand order repaired and checked by reading, not compiled, not run on hardware
//
#include <hip/hip_runtime.h>
#include <hip/hip_bf16.h>
#include <math.h>

typedef _Float16 f16;
typedef __attribute__((ext_vector_type(16))) _Float16 v16h;
typedef __attribute__((ext_vector_type(8)))  float    v8f;
typedef __attribute__((ext_vector_type(4)))  float    v4f;

union Frag { v16h v; v4f f4[2]; };

#define DIMD 512
#define SEQ  2048
#define BATCH 4
#define HEADS 8
#define HDIM 64
#define ROWS (BATCH*SEQ)   // 8192

#if __has_builtin(__builtin_amdgcn_tensor_load_to_lds) && __has_builtin(__builtin_amdgcn_s_wait_tensorcnt)
#define USE_TDM 1
#else
#define USE_TDM 0
#endif

#if USE_TDM
typedef unsigned int v4u_ __attribute__((ext_vector_type(4)));
typedef int          v8i_ __attribute__((ext_vector_type(8)));
typedef int          v4i_ __attribute__((ext_vector_type(4)));

__device__ __forceinline__ unsigned lds_byte_offset(const void* p) {
    // generic -> LDS(addrspace 3) cast; ptrtoint of an AS(3) pointer is the raw LDS byte offset
    return (unsigned)(unsigned long long)(__attribute__((address_space(3))) const void*)p;
}

// 2D tile DMA: global (row-major, row stride = stride_elems f16) -> LDS, with optional
// LDS pad insertion. tile_w elements per row, tile_h rows. pad encoded per D# spec:
// pad_interval: 2^(v+1) DWORDs of data between pads; pad_amount: (v+1) DWORDs inserted.
__device__ __forceinline__ void tdm_load_2d(unsigned lds_off, const f16* gptr,
                                            int dim0, int dim1, int stride_elems,
                                            int tile_w, int tile_h,
                                            int pad_interval, int pad_amount) {
    unsigned long long ga = (unsigned long long)gptr;
    v4u_ g0;
    g0[0] = 1u;                                            // count=1, no gather
    g0[1] = lds_off;                                       // lds_addr
    g0[2] = (unsigned)(ga & 0xffffffffull);                // global_addr[31:0]
    g0[3] = (unsigned)((ga >> 32) & 0x1ffffffull) | (2u << 30);  // addr[56:32] | type=2
    v8i_ g1;
    g1[0] = (1 << 16)                                      // data_size = 2 bytes
          | (1 << 20)                                      // pad_enable
          | (pad_interval << 22) | (pad_amount << 25);
    g1[1] = (dim0 & 0xffff) << 16;                         // tensor_dim0[15:0]
    g1[2] = ((dim0 >> 16) & 0xffff) | ((dim1 & 0xffff) << 16);
    g1[3] = ((dim1 >> 16) & 0xffff) | ((tile_w & 0xffff) << 16);
    g1[4] = tile_h & 0xffff;                               // tile_dim1, tile_dim2=0
    g1[5] = stride_elems;                                  // tensor_dim0_stride[31:0]
    g1[6] = 0;
    g1[7] = 0;
    v4i_ z4 = {0, 0, 0, 0};                                // groups 2/3 unused (2D)
    v8i_ z8 = {0, 0, 0, 0, 0, 0, 0, 0};                    // extra group (6-arg builtin), unused
    __builtin_amdgcn_tensor_load_to_lds(g0, g1, z4, z4, z8, 0);
}
#endif

// ---------------- f32 -> f16 cast ----------------
__global__ void cast_f16_kernel(const float* __restrict__ src, f16* __restrict__ dst, int n) {
    int i = blockIdx.x * 256 + threadIdx.x;
    if (i < n) dst[i] = (f16)src[i];
}

// ---------------- LayerNorm: one wave (32 lanes) per 512-wide row ----------------
template<bool OUT16>
__global__ __launch_bounds__(256) void ln_kernel(const float* __restrict__ x,
                                                 const float* __restrict__ g,
                                                 const float* __restrict__ bta,
                                                 f16* __restrict__ o16,
                                                 float* __restrict__ o32) {
    const int lane = threadIdx.x & 31;
    const int wave = threadIdx.x >> 5;
    const int row  = blockIdx.x * 8 + wave;
    const float* xr = x + (size_t)row * DIMD;
    float vals[16];
    float s = 0.f, s2 = 0.f;
#pragma unroll
    for (int i = 0; i < 4; i++) {
        v4f t = *(const v4f*)(xr + lane * 16 + i * 4);
#pragma unroll
        for (int j = 0; j < 4; j++) { float v = t[j]; vals[i*4+j] = v; s += v; s2 += v*v; }
    }
#pragma unroll
    for (int m = 16; m >= 1; m >>= 1) { s += __shfl_xor(s, m, 32); s2 += __shfl_xor(s2, m, 32); }
    const float mean = s * (1.f/512.f);
    const float var  = s2 * (1.f/512.f) - mean*mean;
    const float rstd = rsqrtf(var + 1e-5f);
#pragma unroll
    for (int i = 0; i < 16; i++) {
        int c = lane * 16 + i;
        float y = (vals[i] - mean) * rstd * g[c] + bta[c];
        if (OUT16) o16[(size_t)row * DIMD + c] = (f16)y;
        else       o32[(size_t)row * DIMD + c] = y;
    }
}

// ---------------- Generic WMMA GEMM: C(M,N) = A(M,K) @ W(N,K)^T + bias, + epilogue ----------------
// MODE 0: out16 = C            (qkv)
// MODE 1: out32 = resid + C    (Wo / W2 residual adds)
// MODE 2: out16 = gelu(C)      (W1)
template<int MODE>
__global__ __launch_bounds__(256) void gemm_kernel(const f16* __restrict__ A,
                                                   const f16* __restrict__ W,
                                                   const float* __restrict__ bias,
                                                   const float* __restrict__ resid,
                                                   f16* __restrict__ out16,
                                                   float* __restrict__ out32,
                                                   int M, int N, int K) {
    __shared__ alignas(16) f16 As[128 * 40];   // [m][k], row stride 40 halves (80B, 16B aligned)
    __shared__ alignas(16) f16 Bs[32 * 136];   // transposed: [k][n], row stride 136 halves (272B)
    const int tid  = threadIdx.x;
    const int lane = tid & 31, wave = tid >> 5;
    const int wm = wave >> 1, wn = wave & 1;   // 4x2 wave grid -> 128x128 block tile
    const int mb0 = blockIdx.y * 128, nb0 = blockIdx.x * 128;
    const int gg = lane >> 4, mm = lane & 15;
#if USE_TDM
    const unsigned as_off = lds_byte_offset(&As[0]);
#endif

    v8f acc[2][4];
#pragma unroll
    for (int i = 0; i < 2; i++)
#pragma unroll
        for (int j = 0; j < 4; j++) acc[i][j] = (v8f){};

    for (int kk = 0; kk < K; kk += 32) {
#if USE_TDM
        // A tile (128 rows x 32 halves, row stride K) via Tensor Data Mover; the TDM's
        // pad-insertion (16 data DWORDs -> +4 pad DWORDs) reproduces the 40-half LDS stride.
        if (wave == 0)
            tdm_load_2d(as_off, A + (size_t)mb0 * K + kk, K, M, K, 32, 128, /*pi=*/3, /*pa=*/3);
#else
        {   // stage A tile (128x32), 2 threads per row
            int row = tid >> 1, col = (tid & 1) * 16;
            const v4f* src = (const v4f*)(A + (size_t)(mb0 + row) * K + kk + col);
            v4f* dst = (v4f*)(As + row * 40 + col);
            dst[0] = src[0]; dst[1] = src[1];
        }
#endif
        // stage W tile transposed into Bs: Bs[k][n] = W[nb0+n][kk+k]  (TDM cannot transpose)
        {
            int n = tid >> 1, c0 = (tid & 1) * 16;
            const f16* src = W + (size_t)(nb0 + n) * K + kk + c0;
            v4f t0 = ((const v4f*)src)[0];
            v4f t1 = ((const v4f*)src)[1];
            const f16* h0 = (const f16*)&t0;
            const f16* h1 = (const f16*)&t1;
#pragma unroll
            for (int j = 0; j < 8; j++) Bs[(c0 + j) * 136 + n] = h0[j];
#pragma unroll
            for (int j = 0; j < 8; j++) Bs[(c0 + 8 + j) * 136 + n] = h1[j];
        }
#if USE_TDM
        __builtin_amdgcn_s_wait_tensorcnt(0);
#endif
        __syncthreads();

        Frag af[2], bf[4];
#pragma unroll
        for (int i = 0; i < 2; i++) {                 // A-matrix layout (16x32 f16)
            const f16* ap = As + (wm * 32 + i * 16 + mm) * 40;
            af[i].f4[0] = *(const v4f*)(ap + 8 * gg);
            af[i].f4[1] = *(const v4f*)(ap + 16 + 8 * gg);
        }
#pragma unroll
        for (int j = 0; j < 4; j++) {                 // B-matrix layout (32x16 f16): lane%16 = N col, elements = K (table order)
            const f16* bp = Bs + wn * 64 + j * 16 + mm;
#pragma unroll
            for (int s = 0; s < 16; s++) bf[j].v[s] = bp[(s < 8 ? 8 * gg + s : 16 + 8 * gg + (s - 8)) * 136];
        }
#pragma unroll
        for (int i = 0; i < 2; i++)
#pragma unroll
            for (int j = 0; j < 4; j++)
                acc[i][j] = __builtin_amdgcn_wmma_f32_16x16x32_f16(
                    false, af[i].v, false, bf[j].v, (short)0, acc[i][j], false, false);
        __syncthreads();
    }

    // epilogue: C/D layout -> row = base + r + 8*(lane/16), col = base + lane%16
#pragma unroll
    for (int i = 0; i < 2; i++)
#pragma unroll
        for (int j = 0; j < 4; j++) {
            int row0 = mb0 + wm * 32 + i * 16 + gg * 8;
            int col  = nb0 + wn * 64 + j * 16 + mm;
            float bv = bias[col];
#pragma unroll
            for (int r = 0; r < 8; r++) {
                size_t idx = (size_t)(row0 + r) * N + col;
                float v = acc[i][j][r] + bv;
                if (MODE == 0) {
                    out16[idx] = (f16)v;
                } else if (MODE == 1) {
                    out32[idx] = resid[idx] + v;
                } else {
                    float u = 0.5f * v * (1.0f + erff(v * 0.70710678118654752f));
                    out16[idx] = (f16)u;
                }
            }
        }
}

// ---------------- Flash attention: 1 workgroup = (batch, head, 64-row Q block), 4 waves ----------------
__global__ __launch_bounds__(128) void attn_kernel(const f16* __restrict__ qkv,
                                                   f16* __restrict__ out) {
    __shared__ alignas(16) f16 Kl[64 * 40];       // transposed: Kl[d][key], stride 40
    __shared__ alignas(16) f16 Vl[32 * 72];       // row-major:  Vl[key][d], stride 72
    __shared__ alignas(16) f16 Pl[4][16 * 40];    // per-wave P staging (C-layout -> A-layout)
    const int tid = threadIdx.x, lane = tid & 31, wave = tid >> 5;
    const int qb = blockIdx.x * 64 + wave * 16;
    const int h = blockIdx.y, b = blockIdx.z;
    const int gg = lane >> 4, mm = lane & 15;
#if USE_TDM
    const unsigned vl_off = lds_byte_offset(&Vl[0]);
#endif

    const f16* Qb = qkv + ((size_t)b * SEQ) * (3 * DIMD) + h * HDIM;
    const f16* Kb = Qb + DIMD;
    const f16* Vb = Qb + 2 * DIMD;

    Frag qf[2];
#pragma unroll
    for (int ks = 0; ks < 2; ks++) {
        const f16* qp = Qb + (size_t)(qb + mm) * (3 * DIMD) + ks * 32 + 8 * gg;
        qf[ks].f4[0] = *(const v4f*)qp;
        qf[ks].f4[1] = *(const v4f*)(qp + 16);
    }

    float mrun[8], lrun[8];
    v8f o[4];
#pragma unroll
    for (int r = 0; r < 8; r++) { mrun[r] = -1e30f; lrun[r] = 0.f; }
#pragma unroll
    for (int n = 0; n < 4; n++) o[n] = (v8f){};

    for (int kb = 0; kb < SEQ; kb += 32) {
#if USE_TDM
        // V tile (32 rows x 64 halves, row stride 1536) via TDM; pad 32 data DWORDs -> +4
        // pad DWORDs reproduces the 72-half LDS stride.
        if (wave == 0)
            tdm_load_2d(vl_off, Vb + (size_t)kb * (3 * DIMD), 3 * DIMD, ROWS, 3 * DIMD,
                        64, 32, /*pi=*/4, /*pa=*/3);
#else
        {   // stage V row-major
            int key = tid >> 2, seg = tid & 3;
            const v4f* vp = (const v4f*)(Vb + (size_t)(kb + key) * (3 * DIMD) + seg * 16);
            v4f* dp = (v4f*)(Vl + key * 72 + seg * 16);
            dp[0] = vp[0]; dp[1] = vp[1];
        }
#endif
        {   // stage K transposed (TDM cannot transpose): key = tid&31, d-chunk = tid>>5
            int key = tid & 31, dg = tid >> 5;
            const f16* kp = Kb + (size_t)(kb + key) * (3 * DIMD) + dg * 16;
            v4f t0 = ((const v4f*)kp)[0];
            v4f t1 = ((const v4f*)kp)[1];
            const f16* h0 = (const f16*)&t0;
            const f16* h1 = (const f16*)&t1;
#pragma unroll
            for (int j = 0; j < 8; j++) Kl[(dg * 16 + j) * 40 + key] = h0[j];
#pragma unroll
            for (int j = 0; j < 8; j++) Kl[(dg * 16 + 8 + j) * 40 + key] = h1[j];
        }
#if USE_TDM
        __builtin_amdgcn_s_wait_tensorcnt(0);
#endif
        __syncthreads();

        // S = Q @ K^T * 1/sqrt(64) : two 16x16 n-tiles, K-dim = head dim (2 steps of 32)
        v8f s[2];
#pragma unroll
        for (int nt = 0; nt < 2; nt++) {
            v8f a = (v8f){};
#pragma unroll
            for (int ks = 0; ks < 2; ks++) {
                Frag bf;
                const f16* bp = Kl + (ks * 32) * 40 + nt * 16 + mm;
#pragma unroll
                for (int s = 0; s < 16; s++) bf.v[s] = bp[(s < 8 ? 8 * gg + s : 16 + 8 * gg + (s - 8)) * 40];
                a = __builtin_amdgcn_wmma_f32_16x16x32_f16(
                        false, qf[ks].v, false, bf.v, (short)0, a, false, false);
            }
            s[nt] = a * 0.125f;
        }

        // online softmax (row reductions across 16-lane groups, xor masks keep lane/16 group)
        float corr[8];
#pragma unroll
        for (int r = 0; r < 8; r++) {
            float t = fmaxf(s[0][r], s[1][r]);
#pragma unroll
            for (int m = 8; m >= 1; m >>= 1) t = fmaxf(t, __shfl_xor(t, m, 32));
            float mn = fmaxf(mrun[r], t);
            corr[r] = expf(mrun[r] - mn);
            mrun[r] = mn;
        }
#pragma unroll
        for (int r = 0; r < 8; r++) {
            float p0 = expf(s[0][r] - mrun[r]);
            float p1 = expf(s[1][r] - mrun[r]);
            s[0][r] = p0; s[1][r] = p1;
            float t = p0 + p1;
#pragma unroll
            for (int m = 8; m >= 1; m >>= 1) t += __shfl_xor(t, m, 32);
            lrun[r] = lrun[r] * corr[r] + t;
        }
#pragma unroll
        for (int n = 0; n < 4; n++)
#pragma unroll
            for (int r = 0; r < 8; r++) o[n][r] *= corr[r];

        // C-layout P -> LDS -> A-layout fragment (same-wave DS ops are in-order)
        f16* pw = &Pl[wave][0];
#pragma unroll
        for (int nt = 0; nt < 2; nt++)
#pragma unroll
            for (int r = 0; r < 8; r++)
                pw[(r + 8 * gg) * 40 + nt * 16 + mm] = (f16)s[nt][r];
        Frag pf;
        pf.f4[0] = *(const v4f*)(pw + mm * 40 + 8 * gg);
        pf.f4[1] = *(const v4f*)(pw + mm * 40 + 16 + 8 * gg);

        // O += P @ V : 4 d-tiles, K-dim = 32 keys (single step)
#pragma unroll
        for (int n = 0; n < 4; n++) {
            Frag vf;
            const f16* vp = Vl + n * 16 + mm;
#pragma unroll
            for (int s = 0; s < 16; s++) vf.v[s] = vp[(s < 8 ? 8 * gg + s : 16 + 8 * gg + (s - 8)) * 72];
            o[n] = __builtin_amdgcn_wmma_f32_16x16x32_f16(
                       false, pf.v, false, vf.v, (short)0, o[n], false, false);
        }
        __syncthreads();
    }

    // normalize + write (B,S,H*hd) as f16 for the Wo GEMM
#pragma unroll
    for (int r = 0; r < 8; r++) lrun[r] = 1.f / lrun[r];
#pragma unroll
    for (int n = 0; n < 4; n++)
#pragma unroll
        for (int r = 0; r < 8; r++) {
            int q = qb + 8 * gg + r;
            int d = n * 16 + mm;
            out[((size_t)(b * SEQ + q)) * DIMD + h * HDIM + d] = (f16)(o[n][r] * lrun[r]);
        }
}

// ---------------- depthwise conv (k=3) + LN + GELU + double residual ----------------
// xbuf <- xbuf + h2 + gelu(ln_c(conv(h2)))
__global__ __launch_bounds__(256) void conv_kernel(float* __restrict__ xbuf,
                                                   const float* __restrict__ h2,
                                                   const float* __restrict__ cw,
                                                   const float* __restrict__ cb,
                                                   const float* __restrict__ lg,
                                                   const float* __restrict__ lb) {
    __shared__ float red[16];
    const int row = blockIdx.x;
    const int s = row & (SEQ - 1);
    const int tid = threadIdx.x, lane = tid & 31, wave = tid >> 5;
    float t[2];
#pragma unroll
    for (int k = 0; k < 2; k++) {
        int c = tid + k * 256;
        float hm = (s > 0)       ? h2[(size_t)(row - 1) * DIMD + c] : 0.f;
        float h0 =                 h2[(size_t)row * DIMD + c];
        float hp = (s < SEQ - 1) ? h2[(size_t)(row + 1) * DIMD + c] : 0.f;
        t[k] = hm * cw[c * 3 + 0] + h0 * cw[c * 3 + 1] + hp * cw[c * 3 + 2] + cb[c];
    }
    float sm = t[0] + t[1], s2 = t[0]*t[0] + t[1]*t[1];
#pragma unroll
    for (int m = 16; m >= 1; m >>= 1) { sm += __shfl_xor(sm, m, 32); s2 += __shfl_xor(s2, m, 32); }
    if (lane == 0) { red[wave] = sm; red[8 + wave] = s2; }
    __syncthreads();
    if (tid == 0) {
        float a = 0.f, a2 = 0.f;
        for (int w = 0; w < 8; w++) { a += red[w]; a2 += red[8 + w]; }
        red[0] = a; red[8] = a2;
    }
    __syncthreads();
    const float mean = red[0] * (1.f/512.f);
    const float var  = red[8] * (1.f/512.f) - mean*mean;
    const float rstd = rsqrtf(var + 1e-5f);
#pragma unroll
    for (int k = 0; k < 2; k++) {
        int c = tid + k * 256;
        size_t idx = (size_t)row * DIMD + c;
        float tn = (t[k] - mean) * rstd * lg[c] + lb[c];
        float ge = 0.5f * tn * (1.0f + erff(tn * 0.70710678118654752f));
        xbuf[idx] = xbuf[idx] + h2[idx] + ge;
    }
}

// ---------------- launch ----------------
extern "C" void kernel_launch(void* const* d_in, const int* in_sizes, int n_in,
                              void* d_out, int out_size, void* d_ws, size_t ws_size,
                              hipStream_t stream) {
    const float* x     = (const float*)d_in[0];
    const float* ln1_g = (const float*)d_in[1];
    const float* ln1_b = (const float*)d_in[2];
    const float* Wqkv  = (const float*)d_in[3];
    const float* bqkv  = (const float*)d_in[4];
    const float* Wo    = (const float*)d_in[5];
    const float* bo    = (const float*)d_in[6];
    const float* ln2_g = (const float*)d_in[7];
    const float* ln2_b = (const float*)d_in[8];
    const float* convw = (const float*)d_in[9];
    const float* convb = (const float*)d_in[10];
    const float* lnc_g = (const float*)d_in[11];
    const float* lnc_b = (const float*)d_in[12];
    const float* ln3_g = (const float*)d_in[13];
    const float* ln3_b = (const float*)d_in[14];
    const float* W1    = (const float*)d_in[15];
    const float* b1    = (const float*)d_in[16];
    const float* W2    = (const float*)d_in[17];
    const float* b2    = (const float*)d_in[18];
    float* out = (float*)d_out;

    char* ws = (char*)d_ws;
    f16*   wqkv16 = (f16*)(ws);                                   // 1536*512
    f16*   wo16   = (f16*)(ws + 1572864);                         // 512*512
    f16*   w116   = (f16*)(ws + 2097152);                         // 2048*512
    f16*   w216   = (f16*)(ws + 4194304);                         // 512*2048
    f16*   bufA   = (f16*)(ws + 6291456);                         // 8192*512 f16 (h1/a/h3)
    f16*   bufB   = (f16*)(ws + 14680064);                        // 8192*2048 f16 (qkv, then m1)
    float* xbuf   = (float*)(ws + 48234496);                      // 8192*512 f32 residual stream
    float* h2buf  = (float*)(ws + 65011712);                      // 8192*512 f32

    // weights -> f16
    cast_f16_kernel<<<(1536*512 + 255)/256, 256, 0, stream>>>(Wqkv, wqkv16, 1536*512);
    cast_f16_kernel<<<( 512*512 + 255)/256, 256, 0, stream>>>(Wo,   wo16,    512*512);
    cast_f16_kernel<<<(2048*512 + 255)/256, 256, 0, stream>>>(W1,   w116,   2048*512);
    cast_f16_kernel<<<( 512*2048+ 255)/256, 256, 0, stream>>>(W2,   w216,    512*2048);

    // h1 = ln1(x) -> f16
    ln_kernel<true><<<ROWS/8, 256, 0, stream>>>(x, ln1_g, ln1_b, bufA, nullptr);
    // qkv = h1 @ Wqkv^T + bqkv -> f16
    gemm_kernel<0><<<dim3(1536/128, ROWS/128), 256, 0, stream>>>(
        bufA, wqkv16, bqkv, nullptr, bufB, nullptr, ROWS, 1536, 512);
    // a = flash_attention(q,k,v) -> f16 (reuses bufA)
    attn_kernel<<<dim3(SEQ/64, HEADS, BATCH), 128, 0, stream>>>(bufB, bufA);
    // x1 = x + a @ Wo^T + bo -> f32
    gemm_kernel<1><<<dim3(512/128, ROWS/128), 256, 0, stream>>>(
        bufA, wo16, bo, x, nullptr, xbuf, ROWS, 512, 512);
    // h2 = ln2(x1) -> f32
    ln_kernel<false><<<ROWS/8, 256, 0, stream>>>(xbuf, ln2_g, ln2_b, nullptr, h2buf);
    // x2 = x1 + h2 + gelu(ln_c(conv(h2)))  (in-place on xbuf)
    conv_kernel<<<ROWS, 256, 0, stream>>>(xbuf, h2buf, convw, convb, lnc_g, lnc_b);
    // h3 = ln3(x2) -> f16 (reuses bufA)
    ln_kernel<true><<<ROWS/8, 256, 0, stream>>>(xbuf, ln3_g, ln3_b, bufA, nullptr);
    // m1 = gelu(h3 @ W1^T + b1) -> f16 (reuses bufB)
    gemm_kernel<2><<<dim3(2048/128, ROWS/128), 256, 0, stream>>>(
        bufA, w116, b1, nullptr, bufB, nullptr, ROWS, 2048, 512);
    // out = x2 + m1 @ W2^T + b2 -> f32
    gemm_kernel<1><<<dim3(512/128, ROWS/128), 256, 0, stream>>>(
        bufB, w216, b2, xbuf, nullptr, out, ROWS, 512, 2048);
}
